// IBRRenderingHead_8211977470291
// MI455X (gfx1250) — matrix operand order repaired and checked by reading, not compiled, not run on hardware
//
#include <hip/hip_runtime.h>
#include <hip/hip_bf16.h>
#include <math.h>

// ---------------------------------------------------------------------------
// IBR rendering head for MI455X (gfx1250, wave32, WMMA + TDM).
// Block = 256 threads (8 waves) <-> 128 elements = 16 pixels x 8 views.
// All >=16-wide linear layers run as v_wmma_f32_16x16x32_f16 tiles with
// activations staged in LDS; weights pre-swizzled into B-fragment layout.
// rgb_feats tile (20KB/block, contiguous) is staged via tensor_load_to_lds.
// ---------------------------------------------------------------------------

typedef __attribute__((ext_vector_type(16))) _Float16 v16h;
typedef __attribute__((ext_vector_type(8)))  _Float16 v8h;
typedef __attribute__((ext_vector_type(8)))  float    v8f;
typedef unsigned int u32x4 __attribute__((ext_vector_type(4)));
typedef int          i32x4 __attribute__((ext_vector_type(4)));
typedef int          i32x8 __attribute__((ext_vector_type(8)));

#define RR 1024
#define SS 64
#define VV 8
#define FF 40
#define EPB 128          // elements per block
#define PPB 16           // pixels per block
#define NTHREADS 256

#if defined(__HIP_DEVICE_COMPILE__) && __has_builtin(__builtin_amdgcn_tensor_load_to_lds)
#define HAVE_TDM 1
#else
#define HAVE_TDM 0
#endif

// ---- LDS carve (bytes) ----------------------------------------------------
constexpr int OFF_OUT    = 0;                         // f32 [128][64] gemm f32 out
constexpr int OFF_X      = OFF_OUT    + 128*64*4;     // f32 [128][32] persistent x
constexpr int OFF_RD     = OFF_X      + 128*32*4;     // f32 [128][4]  ray_diffs
constexpr int OFF_MEAN   = OFF_RD     + 128*4*4;      // f32 [16][40]
constexpr int OFF_VAR    = OFF_MEAN   + 16*40*4;      // f32 [16][40]
constexpr int OFF_SRGB   = OFF_VAR    + 16*40*4;      // f32 [128][3]
constexpr int OFF_W      = OFF_SRGB   + 128*3*4;      // f32 [128]
constexpr int OFF_MASK   = OFF_W      + 128*4;        // f32 [128]
constexpr int OFF_VIS    = OFF_MASK   + 128*4;        // f32 [128]
constexpr int OFF_O      = OFF_VIS    + 128*4;        // f32 [128]
constexpr int OFF_TMP    = OFF_O      + 128*4;        // f32 [128]
constexpr int OFF_INA    = OFF_TMP    + 128*4;        // f16 [128][128] gemm A (big)
// rgb_feats raw tile overlays upper 3/4 of s_inA (dead at that time):
constexpr int OFF_RGBRAW = OFF_INA    + 128*32*2;     // f32 [128][40] = 20480 B
constexpr int OFF_INB    = OFF_INA    + 128*128*2;    // f16 [128][64] gemm f16 out / A
constexpr int OFF_FEATS  = OFF_INB    + 128*64*2;     // f16 [128][40]
constexpr int OFF_WZ_RE1 = OFF_FEATS  + 128*40*2;     // swizzled f16 weights:
constexpr int OFF_WZ_RE2 = OFF_WZ_RE1 + 1024*2;       //   [nt][kc][lane][16]
constexpr int OFF_WZ_B1  = OFF_WZ_RE2 + 1536*2;
constexpr int OFF_WZ_B2  = OFF_WZ_B1  + 8192*2;
constexpr int OFF_WZ_V11 = OFF_WZ_B2  + 2048*2;
constexpr int OFF_WZ_V12 = OFF_WZ_V11 + 1024*2;
constexpr int OFF_WZ_V21 = OFF_WZ_V12 + 1536*2;
constexpr int OFF_WZ_O1  = OFF_WZ_V21 + 1024*2;
constexpr int SMEM_BYTES = OFF_WZ_O1  + 1024*2;       // ~151 KB, < 160 KB (2/WGP)

struct P {
  const float *rgb, *rd, *pm, *ani;
  const float *W_re1,*b_re1,*W_re2,*b_re2,*W_b1,*b_b1,*W_b2,*b_b2;
  const float *W_v11,*b_v11,*W_v12,*b_v12,*W_v21,*b_v21,*W_v22,*b_v22;
  const float *W_o1,*b_o1,*W_o2,*b_o2,*W_o3,*b_o3;
  float* out;
};

__device__ __forceinline__ float elu_f(float x)  { return x > 0.f ? x : (__expf(x) - 1.f); }
__device__ __forceinline__ float sigm_f(float x) { return 1.f / (1.f + __expf(-x)); }

#if HAVE_TDM
// 1-D contiguous TDM: nelem f32 from gptr -> LDS byte offset lds_off.
// D# packing per CDNA5 ISA 8.3/8.4 (group0: count/lds/global/type,
// group1: data_size + tensor_dim0/tile_dim0/stride).
__device__ __forceinline__ void tdm_load_f32_1d(const void* gptr, unsigned lds_off,
                                                unsigned nelem)
{
  unsigned long long ga = (unsigned long long)gptr;
  u32x4 g0;
  g0[0] = 1u;                                          // count=1, user mode
  g0[1] = lds_off;                                     // lds_addr (bytes)
  g0[2] = (unsigned)(ga & 0xffffffffu);                // global_addr[31:0]
  g0[3] = (unsigned)((ga >> 32) & 0x01ffffffu)         // global_addr[56:32]
        | (2u << 30);                                  // type=2 ("image")
  i32x8 g1 = {};
  g1[0] = (int)(2u << 16);                             // data_size=4B, wg_mask=0
  g1[1] = (int)((nelem & 0xffffu) << 16);              // tensor_dim0[15:0] @ bits63:48
  g1[2] = (int)(((nelem >> 16) & 0xffffu)              // tensor_dim0[31:16]
        | (1u << 16));                                 // tensor_dim1 = 1
  g1[3] = (int)((nelem & 0xffffu) << 16);              // tile_dim0 @ bits127:112
  g1[4] = 1;                                           // tile_dim1 = 1
  g1[5] = (int)nelem;                                  // tensor_dim0_stride[31:0]
  g1[6] = 0;                                           // stride[47:32]=0, dim1_stride=0
  g1[7] = 0;
  i32x4 z4a = {}, z4b = {};
#if __clang_major__ >= 23
  i32x8 z8 = {};
  __builtin_amdgcn_tensor_load_to_lds(g0, g1, z4a, z4b, z8, 0);
#else
  __builtin_amdgcn_tensor_load_to_lds(g0, g1, z4a, z4b, 0);
#endif
}
#endif

// Convert W[K][N] (f32, row-major) into WMMA B-fragment-ready f16 layout:
// block b = (nt*kchunks + kc)*32 + lane holds 16 contiguous halfs:
//   n = nt*16 + (lane&15), K = kc*32 + (lane<16 ? 0 : 16) + j, j=0..15
__device__ void load_wswz(const float* __restrict__ Wg, int K, int N,
                          int kchunks, int ntiles, _Float16* __restrict__ dst, int tid)
{
  int blocks = ntiles * kchunks * 32;
  for (int b = tid; b < blocks; b += NTHREADS) {
    int lane = b & 31;
    int g    = b >> 5;
    int kc   = g % kchunks;
    int nt   = g / kchunks;
    int n     = nt*16 + (lane & 15);
    int kbase = kc*32 + ((lane & 16) ? 16 : 0);
    _Float16* d = dst + (size_t)b * 16;
    #pragma unroll
    for (int j = 0; j < 16; ++j) {
      int k = kbase + j;
      float v = (k < K && n < N) ? Wg[k*N + n] : 0.f;
      d[j] = (_Float16)v;
    }
  }
}

// out[128][ldn] = act(in[128][ldk] @ Wswz + bias); 16x16 tiles, K chunks of 32.
// A frag (16x32) uses contiguous k, matching wswz B: lane<16 row M=lane holds K 0-15,
// lane>=16 row M=lane-16 holds K 16-31  -> two ds_load_b128 per frag.
// outHalf=1 writes f16 (feeds next GEMM's A directly, incl. zero-padded cols).
__device__ void gemm_ld(const _Float16* __restrict__ in, int ldk,
                        const _Float16* __restrict__ wswz,
                        void* __restrict__ outp, int ldn, int outHalf,
                        const float* __restrict__ bias, int biasN,
                        int ntiles, int kchunks, int act, int tid)
{
  int wave = tid >> 5, lane = tid & 31;
  int hi   = (lane & 16) ? 16 : 0;     // A half-row select
  int l15  = lane & 15;
  for (int t = wave; t < 8*ntiles; t += 8) {
    int mt = t / ntiles, nt = t - mt*ntiles;
    const _Float16* arow = in + (mt*16 + l15)*ldk + hi;
    const _Float16* wp   = wswz + (size_t)(nt*kchunks*32 + lane) * 16;
    v8f acc = {};
    for (int kc = 0; kc < kchunks; ++kc) {
      v8h a0 = *(const v8h*)(arow + kc*32);
      v8h a1 = *(const v8h*)(arow + kc*32 + 8);
      v16h a = __builtin_shufflevector(a0, a1, 0,1,2,3,4,5,6,7,8,9,10,11,12,13,14,15);
      v8h b0 = *(const v8h*)(wp + kc*512);
      v8h b1 = *(const v8h*)(wp + kc*512 + 8);
      v16h b = __builtin_shufflevector(b0, b1, 0,1,2,3,4,5,6,7,8,9,10,11,12,13,14,15);
      acc = __builtin_amdgcn_wmma_f32_16x16x32_f16(false, a, false, b,
                                                   (short)0, acc, false, false);
    }
    int n  = nt*16 + l15;
    float bv = (n < biasN) ? bias[n] : 0.f;
    int m0 = mt*16 + ((lane & 16) ? 8 : 0);
    #pragma unroll
    for (int i = 0; i < 8; ++i) {          // C layout: VGPR i -> row m0+i
      float v = acc[i] + bv;
      if (act) v = elu_f(v);
      if (outHalf) ((_Float16*)outp)[(m0 + i)*ldn + n] = (_Float16)v;
      else         ((float*)outp)   [(m0 + i)*ldn + n] = v;
    }
  }
}

__global__ __launch_bounds__(NTHREADS)
void ibr_head_kernel(P p)
{
  extern __shared__ char smem[];
  float*    s_out   = (float*)   (smem + OFF_OUT);
  float*    s_x     = (float*)   (smem + OFF_X);
  float*    s_rd    = (float*)   (smem + OFF_RD);
  float*    s_mean  = (float*)   (smem + OFF_MEAN);
  float*    s_var   = (float*)   (smem + OFF_VAR);
  float*    s_srgb  = (float*)   (smem + OFF_SRGB);
  float*    s_w     = (float*)   (smem + OFF_W);
  float*    s_mask  = (float*)   (smem + OFF_MASK);
  float*    s_vis   = (float*)   (smem + OFF_VIS);
  float*    s_o     = (float*)   (smem + OFF_O);
  float*    s_tmp   = (float*)   (smem + OFF_TMP);
  _Float16* s_inA   = (_Float16*)(smem + OFF_INA);
  float*    s_rgbraw= (float*)   (smem + OFF_RGBRAW);
  _Float16* s_inB   = (_Float16*)(smem + OFF_INB);
  _Float16* s_feats = (_Float16*)(smem + OFF_FEATS);
  _Float16* s_wz_re1= (_Float16*)(smem + OFF_WZ_RE1);
  _Float16* s_wz_re2= (_Float16*)(smem + OFF_WZ_RE2);
  _Float16* s_wz_b1 = (_Float16*)(smem + OFF_WZ_B1);
  _Float16* s_wz_b2 = (_Float16*)(smem + OFF_WZ_B2);
  _Float16* s_wz_v11= (_Float16*)(smem + OFF_WZ_V11);
  _Float16* s_wz_v12= (_Float16*)(smem + OFF_WZ_V12);
  _Float16* s_wz_v21= (_Float16*)(smem + OFF_WZ_V21);
  _Float16* s_wz_o1 = (_Float16*)(smem + OFF_WZ_O1);

  const int tid = threadIdx.x;
  const int e0  = blockIdx.x * EPB;
  const int p0  = blockIdx.x * PPB;

  // Kick off async rgb tile -> LDS (TDM); overlaps with weight swizzle +
  // the dir-MLP GEMMs (which only touch the lower 8KB of s_inA).
#if HAVE_TDM
  if (tid < 32)
    tdm_load_f32_1d(p.rgb + (size_t)e0*FF, (unsigned)OFF_RGBRAW, EPB*FF);
#else
  for (int i = tid; i < EPB*FF; i += NTHREADS)
    s_rgbraw[i] = p.rgb[(size_t)e0*FF + i];
#endif
  __builtin_prefetch(p.W_b1, 0, 1);   // global_prefetch_b8 (weights are L2-hot)

  // Stage weights (f32 -> f16, pre-swizzled B fragments).
  load_wswz(p.W_re1,   4, 16, 1, 2, s_wz_re1, tid);   // N padded 16->32: output
  load_wswz(p.W_re2,  16, 40, 1, 3, s_wz_re2, tid);   // tiles directly produce
  load_wswz(p.W_b1,  120, 64, 4, 4, s_wz_b1,  tid);   // next layer's K padding.
  load_wswz(p.W_b2,   64, 32, 2, 2, s_wz_b2,  tid);
  load_wswz(p.W_v11,  32, 32, 1, 2, s_wz_v11, tid);
  load_wswz(p.W_v12,  32, 33, 1, 3, s_wz_v12, tid);
  load_wswz(p.W_v21,  32, 32, 1, 2, s_wz_v21, tid);
  load_wswz(p.W_o1,   37, 16, 2, 1, s_wz_o1,  tid);

  for (int i = tid; i < EPB*4; i += NTHREADS) s_rd[i]   = p.rd[(size_t)e0*4 + i];
  for (int i = tid; i < EPB;   i += NTHREADS) s_mask[i] = p.pm[e0 + i];
  const float al = fabsf(p.ani[0]);
  __syncthreads();

  // ---- dir MLP: 4 -> 16 (elu) -> 40 (elu); K padded to 32 -----------------
  for (int i = tid; i < EPB*32; i += NTHREADS) {
    int e = i >> 5, k = i & 31;
    s_inA[i] = (k < 4) ? (_Float16)s_rd[e*4 + k] : (_Float16)0.f;
  }
  __syncthreads();
  // f16 out, N padded to 32 (cols 16..31 = elu(0) = 0) -> re2's A directly.
  gemm_ld(s_inA, 32, s_wz_re1, s_inB, 32, 1, p.b_re1, 16, 2, 1, 1, tid);
  __syncthreads();
  gemm_ld(s_inB, 32, s_wz_re2, s_out, 48, 0, p.b_re2, 40, 3, 1, 1, tid);
#if HAVE_TDM
  if (tid < 32) __builtin_amdgcn_s_wait_tensorcnt(0);  // rgb tile landed
#endif
  __syncthreads();

  // ---- feats = rgb + dir_feat; src_rgb; view weights ----------------------
  for (int i = tid; i < EPB*FF; i += NTHREADS) {
    int e = i / FF, f = i - e*FF;
    float rv = s_rgbraw[i];
    s_feats[i] = (_Float16)(rv + s_out[e*48 + f]);
    if (f < 3) s_srgb[e*3 + f] = rv;
  }
  if (tid < EPB) s_tmp[tid] = __expf(al * (s_rd[tid*4 + 3] - 1.f));
  __syncthreads();
  if (tid < PPB) {
    int base = tid*VV;
    float mn = s_tmp[base];
    for (int v = 1; v < VV; ++v) mn = fminf(mn, s_tmp[base+v]);
    float wv[VV], sum = 0.f;
    for (int v = 0; v < VV; ++v) { wv[v] = (s_tmp[base+v] - mn)*s_mask[base+v]; sum += wv[v]; }
    float inv = 1.f/(sum + 1e-8f);
    for (int v = 0; v < VV; ++v) s_w[base+v] = wv[v]*inv;
  }
  __syncthreads();

  // ---- weighted mean/var over views ---------------------------------------
  for (int i = tid; i < PPB*FF; i += NTHREADS) {
    int pp = i / FF, f = i - pp*FF;
    int base = pp*VV;
    float m = 0.f;
    for (int v = 0; v < VV; ++v) m += (float)s_feats[(base+v)*FF + f] * s_w[base+v];
    float var = 0.f;
    for (int v = 0; v < VV; ++v) {
      float d = (float)s_feats[(base+v)*FF + f] - m;
      var += s_w[base+v]*d*d;
    }
    s_mean[i] = m; s_var[i] = var;
  }
  __syncthreads();

  // ---- base MLP: concat[mean,var,feats] (120->128 pad) -> 64 -> 32 (x) ----
  for (int i = tid; i < EPB*128; i += NTHREADS) {
    int e = i >> 7, k = i & 127;
    int pp = e >> 3;
    _Float16 v = (_Float16)0.f;
    if      (k < 40)  v = (_Float16)s_mean[pp*FF + k];
    else if (k < 80)  v = (_Float16)s_var[pp*FF + k - 40];
    else if (k < 120) v = s_feats[e*FF + k - 80];
    s_inA[i] = v;
  }
  __syncthreads();
  gemm_ld(s_inA, 128, s_wz_b1, s_inB, 64, 1, p.b_b1, 64, 4, 4, 1, tid); // f16 out
  __syncthreads();
  gemm_ld(s_inB, 64, s_wz_b2, s_x, 32, 0, p.b_b2, 32, 2, 2, 1, tid);
  __syncthreads();

  // ---- pv = elu(lin(elu(lin(x*w, v11)), v12)); res/vis split --------------
  for (int i = tid; i < EPB*32; i += NTHREADS) {
    int e = i >> 5;
    s_inA[i] = (_Float16)(s_x[i]*s_w[e]);
  }
  __syncthreads();
  gemm_ld(s_inA, 32, s_wz_v11, s_inB, 32, 1, p.b_v11, 32, 2, 1, 1, tid); // f16 out
  __syncthreads();
  gemm_ld(s_inB, 32, s_wz_v12, s_out, 48, 0, p.b_v12, 33, 3, 1, 1, tid);
  __syncthreads();
  for (int i = tid; i < EPB*32; i += NTHREADS) {
    int e = i >> 5, k = i & 31;
    s_x[i] += s_out[e*48 + k];
  }
  if (tid < EPB) s_vis[tid] = sigm_f(s_out[tid*48 + 32]);
  __syncthreads();

  // ---- vis head: sigmoid(lin(elu(lin(x*sig(vis)*mask, v21)), v22))*mask ---
  for (int i = tid; i < EPB*32; i += NTHREADS) {
    int e = i >> 5;
    s_inA[i] = (_Float16)(s_x[i]*s_vis[e]*s_mask[e]);
  }
  __syncthreads();
  gemm_ld(s_inA, 32, s_wz_v21, s_out, 32, 0, p.b_v21, 32, 2, 1, 1, tid);
  __syncthreads();
  if (tid < EPB) {
    float acc = p.b_v22[0];
    for (int k = 0; k < 32; ++k) acc += s_out[tid*32 + k]*p.W_v22[k];
    s_vis[tid] = sigm_f(acc)*s_mask[tid];
  }
  __syncthreads();

  // ---- blend head: concat[x,vis,rdiff] (37->64 pad) -> 16 -> 8 -> 1 -------
  for (int i = tid; i < EPB*64; i += NTHREADS) {
    int e = i >> 6, k = i & 63;
    _Float16 v = (_Float16)0.f;
    if      (k < 32)  v = (_Float16)s_x[e*32 + k];
    else if (k == 32) v = (_Float16)s_vis[e];
    else if (k < 37)  v = (_Float16)s_rd[e*4 + (k - 33)];
    s_inA[i] = v;
  }
  __syncthreads();
  gemm_ld(s_inA, 64, s_wz_o1, s_out, 16, 0, p.b_o1, 16, 1, 2, 1, tid);
  __syncthreads();
  if (tid < EPB) {
    float o = p.b_o3[0];
    for (int j = 0; j < 8; ++j) {
      float h = p.b_o2[j];
      for (int k = 0; k < 16; ++k) h += s_out[tid*16 + k]*p.W_o2[k*8 + j];
      o += elu_f(h)*p.W_o3[j];
    }
    s_o[tid] = (s_mask[tid] == 0.f) ? -10000.f : o;
  }
  __syncthreads();

  // ---- softmax over views + blend src_rgb ---------------------------------
  if (tid < PPB) {
    int base = tid*VV;
    float mx = s_o[base];
    for (int v = 1; v < VV; ++v) mx = fmaxf(mx, s_o[base+v]);
    float es[VV], sum = 0.f;
    for (int v = 0; v < VV; ++v) { es[v] = __expf(s_o[base+v] - mx); sum += es[v]; }
    float inv = 1.f/sum;
    float c0 = 0.f, c1 = 0.f, c2 = 0.f;
    for (int v = 0; v < VV; ++v) {
      float bw = es[v]*inv;
      c0 += s_srgb[(base+v)*3 + 0]*bw;
      c1 += s_srgb[(base+v)*3 + 1]*bw;
      c2 += s_srgb[(base+v)*3 + 2]*bw;
    }
    size_t po = ((size_t)p0 + tid)*3;
    p.out[po+0] = c0; p.out[po+1] = c1; p.out[po+2] = c2;
  }
}

extern "C" void kernel_launch(void* const* d_in, const int* in_sizes, int n_in,
                              void* d_out, int out_size, void* d_ws, size_t ws_size,
                              hipStream_t stream)
{
  P p;
  p.rgb = (const float*)d_in[0];
  p.rd  = (const float*)d_in[1];
  p.pm  = (const float*)d_in[2];
  // params dict flattened in jax-pytree (sorted-key) order:
  // W_b1,W_b2,W_o1,W_o2,W_o3,W_re1,W_re2,W_v11,W_v12,W_v21,W_v22,
  // ani_al, b_b1,b_b2,b_o1,b_o2,b_o3,b_re1,b_re2,b_v11,b_v12,b_v21,b_v22
  p.W_b1 =(const float*)d_in[3];  p.W_b2 =(const float*)d_in[4];
  p.W_o1 =(const float*)d_in[5];  p.W_o2 =(const float*)d_in[6];
  p.W_o3 =(const float*)d_in[7];
  p.W_re1=(const float*)d_in[8];  p.W_re2=(const float*)d_in[9];
  p.W_v11=(const float*)d_in[10]; p.W_v12=(const float*)d_in[11];
  p.W_v21=(const float*)d_in[12]; p.W_v22=(const float*)d_in[13];
  p.ani  =(const float*)d_in[14];
  p.b_b1 =(const float*)d_in[15]; p.b_b2 =(const float*)d_in[16];
  p.b_o1 =(const float*)d_in[17]; p.b_o2 =(const float*)d_in[18];
  p.b_o3 =(const float*)d_in[19];
  p.b_re1=(const float*)d_in[20]; p.b_re2=(const float*)d_in[21];
  p.b_v11=(const float*)d_in[22]; p.b_v12=(const float*)d_in[23];
  p.b_v21=(const float*)d_in[24]; p.b_v22=(const float*)d_in[25];
  p.out = (float*)d_out;

  dim3 grid((RR*SS*VV)/EPB), block(NTHREADS);
  ibr_head_kernel<<<grid, block, SMEM_BYTES, stream>>>(p);

  (void)in_sizes; (void)n_in; (void)out_size; (void)d_ws; (void)ws_size;
}
